// ROT_60619168416426
// MI455X (gfx1250) — matrix operand order repaired and checked by reading, not compiled, not run on hardware
//
#include <hip/hip_runtime.h>

#define BB 32
#define NN 1024
#define DD 256
#define EPSF 1e-8f
#define NEGINF -3.402823466e+38f
#define POSINF 3.402823466e+38f

typedef __bf16 bf16_t;
typedef bf16_t v16bf __attribute__((ext_vector_type(16)));
typedef float  v8f   __attribute__((ext_vector_type(8)));

static __device__ __forceinline__ unsigned short f2bf(float f) {
  unsigned int u = __float_as_uint(f);
  u += 0x7FFFu + ((u >> 16) & 1u);
  return (unsigned short)(u >> 16);
}

// ---------------------------------------------------------------------------
// Casts / transposes
// ---------------------------------------------------------------------------
__global__ __launch_bounds__(256) void k_cast_c2(const float* __restrict__ src,
                                                 unsigned short* __restrict__ dst) {
  size_t i = (size_t)blockIdx.x * 256 + threadIdx.x;   // B*N*N threads
  dst[i] = f2bf(src[i]);
}

// dstT[b][c][r] = f( src[b][r][c] );  EXPMASK: f = exp(v)*mask[b*R+r]
template <bool EXPMASK>
__global__ __launch_bounds__(256) void k_castT(const float* __restrict__ src,
                                               unsigned short* __restrict__ dstT,
                                               const float* __restrict__ mask,
                                               int R, int C) {
  __shared__ unsigned short tile[32][33];
  const int b  = blockIdx.z;
  const int c0 = blockIdx.x * 32, r0 = blockIdx.y * 32;
  const int tx = threadIdx.x, ty = threadIdx.y;        // 32 x 8
#pragma unroll
  for (int i = 0; i < 32; i += 8) {
    int r = r0 + ty + i, c = c0 + tx;
    float v = src[((size_t)b * R + r) * C + c];
    if (EXPMASK) v = __expf(v) * mask[(size_t)b * R + r];
    tile[ty + i][tx] = f2bf(v);
  }
  __syncthreads();
#pragma unroll
  for (int i = 0; i < 32; i += 8) {
    int c = c0 + ty + i, r = r0 + tx;
    dstT[((size_t)b * C + c) * R + r] = tile[tx][ty + i];
  }
}

// ---------------------------------------------------------------------------
// Init
// ---------------------------------------------------------------------------
__global__ __launch_bounds__(256) void k_init_big(const float* __restrict__ p0,
                                                  const float* __restrict__ q0,
                                                  const float* __restrict__ mask,
                                                  float* __restrict__ log_t,
                                                  float* __restrict__ log_s,
                                                  float* __restrict__ z) {
  size_t i = (size_t)blockIdx.x * 256 + threadIdx.x;   // B*N*D threads
  size_t row = i / DD;                                 // b*N + n
  size_t b   = row / NN;
  float m  = mask[row];
  float lp = __logf(p0[b * DD + (i % DD)]);
  float lq = __logf(q0[row] + EPSF) * m;
  float v  = (lq + lp) * m;
  log_t[i] = v; log_s[i] = v; z[i] = 0.f;
}

__global__ __launch_bounds__(256) void k_init_nd(const float* __restrict__ q0,
                                                 const float* __restrict__ mask,
                                                 float* __restrict__ log_q0,
                                                 float* __restrict__ log_eta,
                                                 float* __restrict__ z2) {
  size_t i = (size_t)blockIdx.x * 256 + threadIdx.x;   // B*N threads
  float v = __logf(q0[i] + EPSF) * mask[i];
  log_q0[i] = v; log_eta[i] = v; z2[i] = 0.f;
}

__global__ __launch_bounds__(256) void k_init_d(const float* __restrict__ p0,
                                                float* __restrict__ log_p0,
                                                float* __restrict__ log_mu,
                                                float* __restrict__ z1) {
  size_t i = (size_t)blockIdx.x * 256 + threadIdx.x;   // B*D threads
  float v = __logf(p0[i]);
  log_p0[i] = v; log_mu[i] = v; z1[i] = 0.f;
}

// ---------------------------------------------------------------------------
// Batched bf16 WMMA GEMM: C[b] = A[b](MxK) * B[b](KxNcol), B given transposed
// (BT[n][k]).  Wave computes a 32x32 tile via 2x2 v_wmma_f32_16x16x32_bf16.
// Block = 8 waves = 64(M) x 128(Ncol).
// ---------------------------------------------------------------------------
union Frag { v16bf v; uint4 q[2]; };

template <bool OUT_BF16>
__global__ __launch_bounds__(256) void k_gemm(const unsigned short* __restrict__ A,
                                              const unsigned short* __restrict__ BT,
                                              void* __restrict__ C,
                                              int M, int Ncol, int K) {
  const int b    = blockIdx.z;
  const int lane = threadIdx.x & 31;
  const int wave = threadIdx.x >> 5;      // 0..7  -> 2(M) x 4(N)
  const int wm   = wave >> 2;
  const int wn   = wave & 3;
  const int tileM = blockIdx.y * 64 + wm * 32;
  const int tileN = blockIdx.x * 128 + wn * 32;

  const unsigned short* Ab  = A  + (size_t)b * M * K;
  const unsigned short* BTb = BT + (size_t)b * Ncol * K;

  const int l15   = lane & 15;
  const int ahalf = lane >> 4;            // 0/1
  const int akb   = ahalf * 8;            // A: K half offset
  const int bkb   = ahalf * 8;            // B: K half offset

  const unsigned short* arow0 = Ab  + (size_t)(tileM + l15) * K;
  const unsigned short* arow1 = arow0 + (size_t)16 * K;
  const unsigned short* bcol0 = BTb + (size_t)(tileN + l15) * K;
  const unsigned short* bcol1 = bcol0 + (size_t)16 * K;

  v8f acc[2][2] = {};

  for (int k = 0; k < K; k += 32) {
    Frag a0, a1, b0, b1;
    // A 16x32 bf16 layout: lanes 0-15 rows, VGPR0-3 K=kb..kb+7, VGPR4-7 K=kb+16..kb+23
    a0.q[0] = *(const uint4*)(arow0 + k + akb);
    a0.q[1] = *(const uint4*)(arow0 + k + akb + 16);
    a1.q[0] = *(const uint4*)(arow1 + k + akb);
    a1.q[1] = *(const uint4*)(arow1 + k + akb + 16);
    // B 32x16 bf16 layout: lane = column, VGPR0-3 K=kb..kb+7, VGPR4-7 K=kb+16..kb+23
    b0.q[0] = *(const uint4*)(bcol0 + k + bkb);
    b0.q[1] = *(const uint4*)(bcol0 + k + bkb + 16);
    b1.q[0] = *(const uint4*)(bcol1 + k + bkb);
    b1.q[1] = *(const uint4*)(bcol1 + k + bkb + 16);

    acc[0][0] = __builtin_amdgcn_wmma_f32_16x16x32_bf16(false, a0.v, false, b0.v,
                                                        (short)0, acc[0][0], false, false);
    acc[0][1] = __builtin_amdgcn_wmma_f32_16x16x32_bf16(false, a0.v, false, b1.v,
                                                        (short)0, acc[0][1], false, false);
    acc[1][0] = __builtin_amdgcn_wmma_f32_16x16x32_bf16(false, a1.v, false, b0.v,
                                                        (short)0, acc[1][0], false, false);
    acc[1][1] = __builtin_amdgcn_wmma_f32_16x16x32_bf16(false, a1.v, false, b1.v,
                                                        (short)0, acc[1][1], false, false);
  }

  // C/D layout: VGPR r, lane L -> M = r + 8*(L>=16), N = L&15
#pragma unroll
  for (int mt = 0; mt < 2; ++mt)
#pragma unroll
    for (int nt = 0; nt < 2; ++nt)
#pragma unroll
      for (int e = 0; e < 8; ++e) {
        int row = tileM + mt * 16 + e + ahalf * 8;
        int col = tileN + nt * 16 + l15;
        size_t idx = (size_t)b * M * Ncol + (size_t)row * Ncol + col;
        float val = acc[mt][nt][e];
        if (OUT_BF16) ((unsigned short*)C)[idx] = f2bf(val);
        else          ((float*)C)[idx] = val;
      }
}

// ---------------------------------------------------------------------------
// log_t update: y=(x+a0*tmp2-z)/rho+log_s ; log_t = m*(log_eta - lse_d(y)) + y
// also sum_t_row[b][n] = sum_d exp(log_t)*m.  One block per (b,n), 256 thr.
// ---------------------------------------------------------------------------
__global__ __launch_bounds__(256) void k_yt(const float* __restrict__ x,
                                            const float* __restrict__ tmp2,
                                            const float* __restrict__ z,
                                            const float* __restrict__ log_s,
                                            const float* __restrict__ log_eta,
                                            const float* __restrict__ mask,
                                            float* __restrict__ log_t,
                                            float* __restrict__ sum_t_row,
                                            const float* __restrict__ a0a,
                                            const float* __restrict__ rhoa, int it) {
  const int n = blockIdx.x, b = blockIdx.y, d = threadIdx.x;
  const float a0 = a0a[it], rho = rhoa[it];
  const size_t row = (size_t)b * NN + n;
  const size_t i = row * DD + d;
  const float m = mask[row];
  float y = (x[i] + a0 * tmp2[i] - z[i]) / rho + log_s[i];

  __shared__ float red[256];
  red[d] = y; __syncthreads();
  for (int s = 128; s > 0; s >>= 1) { if (d < s) red[d] = fmaxf(red[d], red[d + s]); __syncthreads(); }
  float mx = red[0]; __syncthreads();
  red[d] = __expf(y - mx); __syncthreads();
  for (int s = 128; s > 0; s >>= 1) { if (d < s) red[d] += red[d + s]; __syncthreads(); }
  float lse = __logf(red[0]) + mx; __syncthreads();

  float lt = m * (log_eta[row] - lse) + y;
  log_t[i] = lt;
  red[d] = __expf(lt) * m; __syncthreads();
  for (int s = 128; s > 0; s >>= 1) { if (d < s) red[d] += red[d + s]; __syncthreads(); }
  if (d == 0) sum_t_row[row] = red[0];
}

// ---------------------------------------------------------------------------
// log_s update + z update + column sums. Block owns 32 d-columns of batch b;
// threads (32 x 8), each ty-slice strides over n.
// ---------------------------------------------------------------------------
__global__ __launch_bounds__(256) void k_ys(const float* __restrict__ tmp2,
                                            const float* __restrict__ log_t,
                                            float* __restrict__ z,
                                            float* __restrict__ log_s,
                                            const float* __restrict__ log_mu,
                                            const float* __restrict__ mask,
                                            float* __restrict__ ybuf,
                                            float* __restrict__ sum_s_col,
                                            const float* __restrict__ a0a,
                                            const float* __restrict__ a1a,
                                            const float* __restrict__ rhoa, int it) {
  const int b = blockIdx.y;
  const int tx = threadIdx.x, ty = threadIdx.y;       // 32 x 8
  const int d = blockIdx.x * 32 + tx;
  const float a0 = a0a[it], a1 = a1a[it], rho = rhoa[it];
  const float inv = 1.f / (a1 + rho);

  __shared__ float sA[8][32], sB[8][32], sC[8][32];

  // pass 1: y, min, masked/unmasked max
  float vmin = POSINF, vmax1 = NEGINF, vmax0 = NEGINF;
  for (int n = ty; n < NN; n += 8) {
    size_t row = (size_t)b * NN + n;
    size_t i = row * DD + d;
    float y = (z[i] + a0 * tmp2[i] + rho * log_t[i]) * inv;
    ybuf[i] = y;
    float m = mask[row];
    vmin = fminf(vmin, y);
    if (m != 0.f) vmax1 = fmaxf(vmax1, y); else vmax0 = fmaxf(vmax0, y);
  }
  sA[ty][tx] = vmin; sB[ty][tx] = vmax1; sC[ty][tx] = vmax0;
  __syncthreads();
  if (ty == 0) {
#pragma unroll
    for (int j = 1; j < 8; ++j) {
      vmin = fminf(vmin, sA[j][tx]); vmax1 = fmaxf(vmax1, sB[j][tx]); vmax0 = fmaxf(vmax0, sC[j][tx]);
    }
    sA[0][tx] = fmaxf(vmax1, vmin + vmax0);           // ymax (stab_max)
  }
  __syncthreads();
  const float ymax = sA[0][tx];

  // pass 2: sum exp((y-ymax)*m)*m over n
  float ssum = 0.f;
  for (int n = ty; n < NN; n += 8) {
    size_t row = (size_t)b * NN + n;
    size_t i = row * DD + d;
    float m = mask[row];
    ssum += __expf((ybuf[i] - ymax) * m) * m;
  }
  __syncthreads();
  sB[ty][tx] = ssum; __syncthreads();
  if (ty == 0) {
#pragma unroll
    for (int j = 1; j < 8; ++j) ssum += sB[j][tx];
    sB[0][tx] = __logf(ssum);
  }
  __syncthreads();
  const float lcs = sB[0][tx];
  const float lmu = log_mu[(size_t)b * DD + d];

  // pass 3: write log_s, update z, accumulate sum_s
  float sacc = 0.f;
  for (int n = ty; n < NN; n += 8) {
    size_t row = (size_t)b * NN + n;
    size_t i = row * DD + d;
    float m  = mask[row];
    float ls = m * (lmu - lcs - ymax) + ybuf[i];
    log_s[i] = ls;
    float s_el = __expf(ls) * m;
    float t_el = __expf(log_t[i]) * m;
    z[i] += rho * (t_el - s_el);
    sacc += s_el;
  }
  __syncthreads();
  sC[ty][tx] = sacc; __syncthreads();
  if (ty == 0) {
#pragma unroll
    for (int j = 1; j < 8; ++j) sacc += sC[j][tx];
    sum_s_col[(size_t)b * DD + d] = sacc;
  }
}

// ---------------------------------------------------------------------------
// mu update + z1.  One block per batch, 256 threads over D.
// ---------------------------------------------------------------------------
__global__ __launch_bounds__(256) void k_mu(float* __restrict__ log_mu,
                                            float* __restrict__ z1,
                                            const float* __restrict__ log_p0,
                                            const float* __restrict__ sum_s_col,
                                            const float* __restrict__ a2a,
                                            const float* __restrict__ rhoa, int it) {
  const int b = blockIdx.x, d = threadIdx.x;
  const float a2 = a2a[it], rho = rhoa[it];
  const size_t i = (size_t)b * DD + d;
  float y = (rho * log_mu[i] + a2 * log_p0[i] - z1[i]) / (rho + a2);

  __shared__ float red[256];
  red[d] = y; __syncthreads();
  for (int s = 128; s > 0; s >>= 1) { if (d < s) red[d] = fmaxf(red[d], red[d + s]); __syncthreads(); }
  float mx = red[0]; __syncthreads();
  red[d] = __expf(y - mx); __syncthreads();
  for (int s = 128; s > 0; s >>= 1) { if (d < s) red[d] += red[d + s]; __syncthreads(); }
  float lse = __logf(red[0]) + mx;

  float lm = y - lse;
  log_mu[i] = lm;
  z1[i] += rho * (__expf(lm) - sum_s_col[i]);
}

// ---------------------------------------------------------------------------
// eta update + z2.  One block per batch, 256 threads, 4 n-values each.
// ---------------------------------------------------------------------------
__global__ __launch_bounds__(256) void k_eta(float* __restrict__ log_eta,
                                             float* __restrict__ z2,
                                             const float* __restrict__ log_q0,
                                             const float* __restrict__ sum_t_row,
                                             const float* __restrict__ mask,
                                             const float* __restrict__ a3a,
                                             const float* __restrict__ rhoa, int it) {
  const int b = blockIdx.x, t = threadIdx.x;
  const float a3 = a3a[it], rho = rhoa[it];
  const float inv = 1.f / (rho + a3);

  float yv[4], m[4];
  float vmin = POSINF, vmax1 = NEGINF, vmax0 = NEGINF;
#pragma unroll
  for (int j = 0; j < 4; ++j) {
    size_t i = (size_t)b * NN + t + j * 256;
    m[j] = mask[i];
    yv[j] = (rho * log_eta[i] + a3 * log_q0[i] - z2[i]) * inv * m[j];
    vmin = fminf(vmin, yv[j]);
    if (m[j] != 0.f) vmax1 = fmaxf(vmax1, yv[j]); else vmax0 = fmaxf(vmax0, yv[j]);
  }
  __shared__ float r1[256], r2[256], r3[256];
  r1[t] = vmin; r2[t] = vmax1; r3[t] = vmax0; __syncthreads();
  for (int s = 128; s > 0; s >>= 1) {
    if (t < s) {
      r1[t] = fminf(r1[t], r1[t + s]);
      r2[t] = fmaxf(r2[t], r2[t + s]);
      r3[t] = fmaxf(r3[t], r3[t + s]);
    }
    __syncthreads();
  }
  const float ymax = fmaxf(r2[0], r1[0] + r3[0]);
  __syncthreads();

  float ssum = 0.f;
#pragma unroll
  for (int j = 0; j < 4; ++j) ssum += __expf((yv[j] - ymax) * m[j]) * m[j];
  r1[t] = ssum; __syncthreads();
  for (int s = 128; s > 0; s >>= 1) { if (t < s) r1[t] += r1[t + s]; __syncthreads(); }
  const float lcs = __logf(r1[0]);

#pragma unroll
  for (int j = 0; j < 4; ++j) {
    size_t i = (size_t)b * NN + t + j * 256;
    float le = (yv[j] - lcs - ymax) * m[j];
    log_eta[i] = le;
    z2[i] += rho * (__expf(le) * m[j] - sum_t_row[i]) * m[j];
  }
}

__global__ __launch_bounds__(256) void k_out(const float* __restrict__ log_t,
                                             const float* __restrict__ mask,
                                             float* __restrict__ out) {
  size_t i = (size_t)blockIdx.x * 256 + threadIdx.x;
  out[i] = __expf(log_t[i]) * mask[i / DD];
}

// ---------------------------------------------------------------------------
extern "C" void kernel_launch(void* const* d_in, const int* in_sizes, int n_in,
                              void* d_out, int out_size, void* d_ws, size_t ws_size,
                              hipStream_t stream) {
  const float* x    = (const float*)d_in[0];
  const float* c1   = (const float*)d_in[1];
  const float* c2   = (const float*)d_in[2];
  const float* p0   = (const float*)d_in[3];
  const float* q0   = (const float*)d_in[4];
  const float* a0   = (const float*)d_in[5];
  const float* a1   = (const float*)d_in[6];
  const float* a2   = (const float*)d_in[7];
  const float* a3   = (const float*)d_in[8];
  const float* rho  = (const float*)d_in[9];
  const float* mask = (const float*)d_in[10];

  char* w = (char*)d_ws;
  auto carve = [&](size_t bytes) -> void* {
    void* p = (void*)w;
    w += (bytes + 255) & ~(size_t)255;
    return p;
  };
  unsigned short* c2b   = (unsigned short*)carve((size_t)BB * NN * NN * 2); // 64 MB
  unsigned short* c1bT  = (unsigned short*)carve((size_t)BB * DD * DD * 2); //  4 MB
  unsigned short* SbT   = (unsigned short*)carve((size_t)BB * DD * NN * 2); // 16 MB
  unsigned short* Pb    = (unsigned short*)carve((size_t)BB * NN * DD * 2); // 16 MB
  float* tmp2    = (float*)carve((size_t)BB * NN * DD * 4);                 // 32 MB
  float* ybuf    = (float*)carve((size_t)BB * NN * DD * 4);                 // 32 MB
  float* log_t   = (float*)carve((size_t)BB * NN * DD * 4);
  float* log_s   = (float*)carve((size_t)BB * NN * DD * 4);
  float* zbuf    = (float*)carve((size_t)BB * NN * DD * 4);
  float* log_p0  = (float*)carve((size_t)BB * DD * 4);
  float* log_q0  = (float*)carve((size_t)BB * NN * 4);
  float* log_mu  = (float*)carve((size_t)BB * DD * 4);
  float* log_eta = (float*)carve((size_t)BB * NN * 4);
  float* z1      = (float*)carve((size_t)BB * DD * 4);
  float* z2      = (float*)carve((size_t)BB * NN * 4);
  float* sum_s_col = (float*)carve((size_t)BB * DD * 4);
  float* sum_t_row = (float*)carve((size_t)BB * NN * 4);

  // one-time casts + init
  k_cast_c2<<<(BB * NN * NN) / 256, 256, 0, stream>>>(c2, c2b);
  k_castT<false><<<dim3(DD / 32, DD / 32, BB), dim3(32, 8), 0, stream>>>(c1, c1bT, mask, DD, DD);
  k_init_big<<<(BB * NN * DD) / 256, 256, 0, stream>>>(p0, q0, mask, log_t, log_s, zbuf);
  k_init_nd<<<(BB * NN) / 256, 256, 0, stream>>>(q0, mask, log_q0, log_eta, z2);
  k_init_d<<<(BB * DD) / 256, 256, 0, stream>>>(p0, log_p0, log_mu, z1);

  const dim3 gemmGrid(DD / 128, NN / 64, BB);
  const dim3 trGrid(DD / 32, NN / 32, BB);
  const dim3 trBlk(32, 8);

  for (int k = 0; k < 4; ++k) {
    // tmp2 = c2 @ exp(log_s)*mask @ c1
    k_castT<true><<<trGrid, trBlk, 0, stream>>>(log_s, SbT, mask, NN, DD);
    k_gemm<true ><<<gemmGrid, 256, 0, stream>>>(c2b, SbT, Pb, NN, DD, NN);
    k_gemm<false><<<gemmGrid, 256, 0, stream>>>(Pb, c1bT, tmp2, NN, DD, DD);
    // log_t update + row sums of t
    k_yt<<<dim3(NN, BB), 256, 0, stream>>>(x, tmp2, zbuf, log_s, log_eta, mask,
                                           log_t, sum_t_row, a0, rho, k);
    // tmp2 = c2 @ exp(log_t)*mask @ c1
    k_castT<true><<<trGrid, trBlk, 0, stream>>>(log_t, SbT, mask, NN, DD);
    k_gemm<true ><<<gemmGrid, 256, 0, stream>>>(c2b, SbT, Pb, NN, DD, NN);
    k_gemm<false><<<gemmGrid, 256, 0, stream>>>(Pb, c1bT, tmp2, NN, DD, DD);
    // log_s update + z update + col sums of s
    k_ys<<<dim3(DD / 32, BB), dim3(32, 8), 0, stream>>>(tmp2, log_t, zbuf, log_s, log_mu,
                                                        mask, ybuf, sum_s_col, a0, a1, rho, k);
    // mu / eta / z1 / z2
    k_mu<<<BB, 256, 0, stream>>>(log_mu, z1, log_p0, sum_s_col, a2, rho, k);
    k_eta<<<BB, 256, 0, stream>>>(log_eta, z2, log_q0, sum_t_row, mask, a3, rho, k);
  }

  k_out<<<(BB * NN * DD) / 256, 256, 0, stream>>>(log_t, mask, (float*)d_out);
}
